// ConvLayer_50869592655544
// MI455X (gfx1250) — matrix operand order repaired and checked by reading, not compiled, not run on hardware
//
#include <hip/hip_runtime.h>
#include <hip/hip_bf16.h>

// ---------------------------------------------------------------------------
// Types for CDNA5 WMMA
// ---------------------------------------------------------------------------
typedef __attribute__((ext_vector_type(16))) __bf16 v16bf;
typedef __attribute__((ext_vector_type(8)))  __bf16 v8bf;
typedef __attribute__((ext_vector_type(8)))  float  v8f;

__device__ __forceinline__ __bf16 f32_to_bf16(float f) {
  unsigned u = __builtin_bit_cast(unsigned, f);
  unsigned r = (u + 0x7fffu + ((u >> 16) & 1u)) >> 16;
  return __builtin_bit_cast(__bf16, (unsigned short)r);
}

// A fragment: lane holds two 8-elem chunks at +0 and +16 (K-space)
__device__ __forceinline__ v16bf ld_frag_a(const __bf16* p) {
  v8bf x0 = *reinterpret_cast<const v8bf*>(p);
  v8bf x1 = *reinterpret_cast<const v8bf*>(p + 16);
  v16bf r;
#pragma unroll
  for (int t = 0; t < 8; ++t) { r[t] = x0[t]; r[t + 8] = x1[t]; }
  return r;
}
// B fragment: lane holds two 8-elem chunks at +0 and +16 (K-space)
__device__ __forceinline__ v16bf ld_frag_b(const __bf16* p) {
  v8bf x0 = *reinterpret_cast<const v8bf*>(p);
  v8bf x1 = *reinterpret_cast<const v8bf*>(p + 16);
  v16bf r;
#pragma unroll
  for (int t = 0; t < 8; ++t) { r[t] = x0[t]; r[t + 8] = x1[t]; }
  return r;
}

// ---------------------------------------------------------------------------
// Problem constants
// ---------------------------------------------------------------------------
#define NN    25000      // nodes
#define NE2   100000     // 2*edges (edge rows)
#define NC5R  40000      // 5*N_C5
#define NC6R  48000      // 6*N_C6
#define NCR   88000      // cycle rows

// ---------------------------------------------------------------------------
// Scatter / build kernels (f32 atomics; indices are algebraically fused)
// ---------------------------------------------------------------------------
__global__ void k_hist(const int* __restrict__ idx, int* __restrict__ deg, int n) {
  int i = blockIdx.x * blockDim.x + threadIdx.x;
  if (i < n) atomicAdd(&deg[idx[i]], 1);
}

__global__ void k_scatter_rows(const float* __restrict__ src, const int* __restrict__ idx,
                               float* __restrict__ dst, int D) {
  int r = blockIdx.x;
  int d = idx[r];
  const float* s = src + (size_t)r * D;
  float* o = dst + (size_t)d * D;
  for (int c = threadIdx.x; c < D; c += blockDim.x) atomicAdd(&o[c], s[c]);
}

__global__ void k_scatter_nacc2(const float* __restrict__ nacc1, const int* __restrict__ ea,
                                float* __restrict__ nacc2) {
  int i = blockIdx.x;
  int a = ea[i], p = ea[i ^ 1];
  const float* ra = nacc1 + (size_t)a * 128;
  const float* rp = nacc1 + (size_t)p * 128;
  float* dst = nacc2 + (size_t)a * 256;
  for (int c = threadIdx.x; c < 256; c += blockDim.x) {
    float v = (c < 128) ? ra[c] : (ra[c - 128] + rp[c - 128]);
    atomicAdd(&dst[c], v);
  }
}

__global__ void k_scatter_e2n(const float* __restrict__ nacc2, const int* __restrict__ ea,
                              const int* __restrict__ deg, const float* __restrict__ node_rep,
                              float* __restrict__ e2n) {
  int i = blockIdx.x;
  int a = ea[i], p = ea[i ^ 1];
  float da = (float)deg[a], dp = (float)deg[p];
  float* dst = e2n + (size_t)a * 768;
  for (int c = threadIdx.x; c < 768; c += blockDim.x) {
    float v;
    if (c < 256)      v = nacc2[(size_t)a * 256 + c];
    else if (c < 512) v = nacc2[(size_t)a * 256 + (c - 256)] + nacc2[(size_t)p * 256 + (c - 256)];
    else if (c < 640) v = da * node_rep[(size_t)a * 128 + (c - 512)];
    else              v = da * node_rep[(size_t)a * 128 + (c - 640)]
                        + dp * node_rep[(size_t)p * 128 + (c - 640)];
    atomicAdd(&dst[c], v);
  }
}

__global__ void k_build_Xee(const float* __restrict__ edge_rep, const float* __restrict__ nacc2,
                            const int* __restrict__ ea, const int* __restrict__ deg,
                            const float* __restrict__ node_rep, __bf16* __restrict__ X) {
  int i = blockIdx.x;
  int a = ea[i], p = ea[i ^ 1];
  float da = (float)deg[a], dp = (float)deg[p];
  __bf16* dst = X + (size_t)i * 896;
  for (int c = threadIdx.x; c < 896; c += blockDim.x) {
    float v;
    if (c < 128)      v = edge_rep[(size_t)i * 128 + c];
    else if (c < 384) v = nacc2[(size_t)a * 256 + (c - 128)];
    else if (c < 640) v = nacc2[(size_t)a * 256 + (c - 384)] + nacc2[(size_t)p * 256 + (c - 384)];
    else if (c < 768) v = da * node_rep[(size_t)a * 128 + (c - 640)];
    else              v = da * node_rep[(size_t)a * 128 + (c - 768)]
                        + dp * node_rep[(size_t)p * 128 + (c - 768)];
    dst[c] = f32_to_bf16(v);
  }
}

__global__ void k_build_Xen(const float* __restrict__ node_rep, const float* __restrict__ e2n,
                            __bf16* __restrict__ X) {
  int r = blockIdx.x;
  __bf16* dst = X + (size_t)r * 896;
  for (int c = threadIdx.x; c < 896; c += blockDim.x) {
    float v = (c < 128) ? node_rep[(size_t)r * 128 + c] : e2n[(size_t)r * 768 + (c - 128)];
    dst[c] = f32_to_bf16(v);
  }
}

__global__ void k_scatter_ss1(const float* __restrict__ nacc1, const int* __restrict__ atoms,
                              float* __restrict__ ss, int per) {
  int j = blockIdx.x;
  int a = atoms[j], s = j / per;
  const float* src = nacc1 + (size_t)a * 128;
  float* dst = ss + (size_t)s * 128;
  for (int c = threadIdx.x; c < 128; c += blockDim.x) atomicAdd(&dst[c], src[c]);
}

__global__ void k_scatter_naccc(const float* __restrict__ nacc1, const float* __restrict__ ss,
                                const int* __restrict__ atoms, float* __restrict__ naccc, int per) {
  int j = blockIdx.x;
  int a = atoms[j], s = j / per;
  float* dst = naccc + (size_t)a * 256;
  for (int c = threadIdx.x; c < 256; c += blockDim.x) {
    float v = (c < 128) ? nacc1[(size_t)a * 128 + c] : ss[(size_t)s * 128 + (c - 128)];
    atomicAdd(&dst[c], v);
  }
}

__global__ void k_scatter_ssp(const float* __restrict__ naccc, const int* __restrict__ atoms,
                              float* __restrict__ ssp, int per) {
  int j = blockIdx.x;
  int a = atoms[j], s = j / per;
  const float* src = naccc + (size_t)a * 256;
  float* dst = ssp + (size_t)s * 256;
  for (int c = threadIdx.x; c < 256; c += blockDim.x) atomicAdd(&dst[c], src[c]);
}

__global__ void k_build_cycle(const float* __restrict__ naccc5, const float* __restrict__ ssp5,
                              const float* __restrict__ naccc6, const float* __restrict__ ssp6,
                              const float* __restrict__ cycle_rep, const int* __restrict__ cyatom,
                              __bf16* __restrict__ Xcy, float* __restrict__ nacc_c) {
  int j = blockIdx.x;  // 0..87999
  const float* naccc; const float* ssp; int s;
  if (j < NC5R) { naccc = naccc5; ssp = ssp5; s = j / 5; }
  else          { naccc = naccc6; ssp = ssp6; s = (j - NC5R) / 6; }
  int a = cyatom[j];
  __bf16* xd = Xcy + (size_t)j * 640;
  float*  nd = nacc_c + (size_t)a * 640;
  for (int c = threadIdx.x; c < 640; c += blockDim.x) {
    float v;
    if (c < 256)      v = naccc[(size_t)a * 256 + c];
    else if (c < 512) v = ssp[(size_t)s * 256 + (c - 256)];
    else              v = cycle_rep[(size_t)j * 128 + (c - 512)];
    xd[c] = f32_to_bf16(v);
    atomicAdd(&nd[c], v);
  }
}

__global__ void k_build_Xec(const float* __restrict__ edge_rep, const float* __restrict__ nacc_c,
                            const int* __restrict__ ea, __bf16* __restrict__ X) {
  int i = blockIdx.x;
  int a = ea[i], p = ea[i ^ 1];
  __bf16* dst = X + (size_t)i * 1408;
  for (int c = threadIdx.x; c < 1408; c += blockDim.x) {
    float v;
    if (c < 128)      v = edge_rep[(size_t)i * 128 + c];
    else if (c < 768) v = nacc_c[(size_t)a * 640 + (c - 128)];
    else              v = nacc_c[(size_t)a * 640 + (c - 768)] + nacc_c[(size_t)p * 640 + (c - 768)];
    dst[c] = f32_to_bf16(v);
  }
}

__global__ void k_wt(const float* __restrict__ W, __bf16* __restrict__ Wt, int K, int N) {
  int t = blockIdx.x * blockDim.x + threadIdx.x;
  if (t >= K * N) return;
  int k = t / N, n = t % N;
  Wt[(size_t)n * K + k] = f32_to_bf16(W[(size_t)k * N + n]);
}

// ---------------------------------------------------------------------------
// WMMA GEMM: Y[M,N] = X[M,K](bf16) @ Wt[N,K]^T(bf16), f32 accum + BN stats.
// 8 waves/block arranged 4(M) x 2(N); each wave owns a 32x32 tile (2x2
// fragments) -> each A/B fragment is reused by two wmmas (2 loads per wmma).
// Out-of-range M rows are CLAMPED on load: output row m depends only on
// A row m, so garbage rows are simply excluded from stores/stats.
// K % 32 == 0, N % 64 == 0.
// ---------------------------------------------------------------------------
__global__ __launch_bounds__(256) void k_gemm_bn(
    const __bf16* __restrict__ X, const __bf16* __restrict__ Wt,
    float* __restrict__ Y, float* __restrict__ colsum, float* __restrict__ colsumsq,
    int M, int N, int K) {
  const int wave = threadIdx.x >> 5;
  const int lane = threadIdx.x & 31;
  const int wm = wave & 3;         // 0..3 -> M
  const int wn = wave >> 2;        // 0..1 -> N
  const int m0 = blockIdx.x * 128 + wm * 32;
  const int n0 = blockIdx.y * 64 + wn * 32;
  if (m0 >= M) return;             // wave-uniform exit

  const int lhalf = lane >> 4;     // 0: lanes 0-15, 1: lanes 16-31
  const int lrow  = lane & 15;
  const int aoff  = lhalf * 8;     // A chunk base within K-step
  const int boff  = lhalf * 8;     // B chunk base within K-step

  int am0 = m0 + lrow;       if (am0 >= M) am0 = M - 1;   // clamp, see note
  int am1 = m0 + 16 + lrow;  if (am1 >= M) am1 = M - 1;
  const __bf16* arow0 = X + (size_t)am0 * K + aoff;
  const __bf16* arow1 = X + (size_t)am1 * K + aoff;
  const __bf16* brow0 = Wt + (size_t)(n0 + lrow) * K + boff;
  const __bf16* brow1 = Wt + (size_t)(n0 + 16 + lrow) * K + boff;

  v8f c00 = {}, c01 = {}, c10 = {}, c11 = {};
  for (int k0 = 0; k0 < K; k0 += 32) {
    v16bf a0 = ld_frag_a(arow0 + k0);
    v16bf a1 = ld_frag_a(arow1 + k0);
    v16bf b0 = ld_frag_b(brow0 + k0);
    v16bf b1 = ld_frag_b(brow1 + k0);
    c00 = __builtin_amdgcn_wmma_f32_16x16x32_bf16(false, a0, false, b0, (short)0, c00, false, false);
    c01 = __builtin_amdgcn_wmma_f32_16x16x32_bf16(false, a0, false, b1, (short)0, c01, false, false);
    c10 = __builtin_amdgcn_wmma_f32_16x16x32_bf16(false, a1, false, b0, (short)0, c10, false, false);
    c11 = __builtin_amdgcn_wmma_f32_16x16x32_bf16(false, a1, false, b1, (short)0, c11, false, false);
  }

  // Store D tiles + accumulate BN column statistics.
  const int nA = n0 + lrow;        // column for *0 tiles
  const int nB = n0 + 16 + lrow;   // column for *1 tiles
#pragma unroll
  for (int mt = 0; mt < 2; ++mt) {
    const int mbase = m0 + mt * 16 + lhalf * 8;
    v8f cA = mt ? c10 : c00;
    v8f cB = mt ? c11 : c01;
    float sA = 0.f, qA = 0.f, sB = 0.f, qB = 0.f;
#pragma unroll
    for (int r = 0; r < 8; ++r) {
      int m = mbase + r;
      if (m < M) {
        float vA = cA[r], vB = cB[r];
        Y[(size_t)m * N + nA] = vA;
        Y[(size_t)m * N + nB] = vB;
        sA += vA; qA += vA * vA;
        sB += vB; qB += vB * vB;
      }
    }
    sA += __shfl_down(sA, 16, 32); qA += __shfl_down(qA, 16, 32);
    sB += __shfl_down(sB, 16, 32); qB += __shfl_down(qB, 16, 32);
    if (lhalf == 0) {
      atomicAdd(&colsum[nA], sA);  atomicAdd(&colsumsq[nA], qA);
      atomicAdd(&colsum[nB], sB);  atomicAdd(&colsumsq[nB], qB);
    }
  }
}

// ---------------------------------------------------------------------------
// BatchNorm + ReLU normalizers
// ---------------------------------------------------------------------------
__global__ void k_bn_relu_bf16(const float* __restrict__ Y, const float* __restrict__ colsum,
                               const float* __restrict__ colsumsq, const float* __restrict__ g,
                               const float* __restrict__ bb, __bf16* __restrict__ out,
                               int M, int N, int ldo, int coloff) {
  int row = blockIdx.x;
  for (int n = threadIdx.x; n < N; n += blockDim.x) {
    float mean = colsum[n] / (float)M;
    float var  = colsumsq[n] / (float)M - mean * mean;
    float a  = g[n] * rsqrtf(var + 1e-5f);
    float c0 = bb[n] - mean * a;
    float v = fmaxf(a * Y[(size_t)row * N + n] + c0, 0.f);
    out[(size_t)row * ldo + coloff + n] = f32_to_bf16(v);
  }
}

__global__ void k_bn_relu_res(const float* __restrict__ Y, const float* __restrict__ colsum,
                              const float* __restrict__ colsumsq, const float* __restrict__ g,
                              const float* __restrict__ bb, const float* __restrict__ res,
                              float* __restrict__ out, int M, int N) {
  int row = blockIdx.x;
  for (int n = threadIdx.x; n < N; n += blockDim.x) {
    float mean = colsum[n] / (float)M;
    float var  = colsumsq[n] / (float)M - mean * mean;
    float a  = g[n] * rsqrtf(var + 1e-5f);
    float c0 = bb[n] - mean * a;
    float v = fmaxf(a * Y[(size_t)row * N + n] + c0, 0.f);
    out[(size_t)row * N + n] = res[(size_t)row * N + n] + v;
  }
}

// ---------------------------------------------------------------------------
// Host orchestration
// ---------------------------------------------------------------------------
extern "C" void kernel_launch(void* const* d_in, const int* in_sizes, int n_in,
                              void* d_out, int out_size, void* d_ws, size_t ws_size,
                              hipStream_t stream) {
  (void)in_sizes; (void)n_in; (void)out_size; (void)ws_size;
  const float* node_rep  = (const float*)d_in[0];
  const float* edge_rep  = (const float*)d_in[1];
  const float* cycle_rep = (const float*)d_in[2];
  const int*   edge_atom = (const int*)d_in[3];
  const int*   cyc_atom  = (const int*)d_in[4];
  const float* enn_W1 = (const float*)d_in[5];  const float* enn_g1 = (const float*)d_in[6];
  const float* enn_b1 = (const float*)d_in[7];  const float* enn_W2 = (const float*)d_in[8];
  const float* enn_g2 = (const float*)d_in[9];  const float* enn_b2 = (const float*)d_in[10];
  const float* ene_W1 = (const float*)d_in[11]; const float* ene_g1 = (const float*)d_in[12];
  const float* ene_b1 = (const float*)d_in[13]; const float* ene_W2 = (const float*)d_in[14];
  const float* ene_g2 = (const float*)d_in[15]; const float* ene_b2 = (const float*)d_in[16];
  const float* ece_W1 = (const float*)d_in[17]; const float* ece_g1 = (const float*)d_in[18];
  const float* ece_b1 = (const float*)d_in[19]; const float* ece_W2 = (const float*)d_in[20];
  const float* ece_g2 = (const float*)d_in[21]; const float* ece_b2 = (const float*)d_in[22];
  const float* ecc_W1 = (const float*)d_in[23]; const float* ecc_g1 = (const float*)d_in[24];
  const float* ecc_b1 = (const float*)d_in[25]; const float* ecc_W2 = (const float*)d_in[26];
  const float* ecc_g2 = (const float*)d_in[27]; const float* ecc_b2 = (const float*)d_in[28];
  const float* out_W  = (const float*)d_in[29]; const float* out_g  = (const float*)d_in[30];
  const float* out_b  = (const float*)d_in[31];

  float* out = (float*)d_out;
  float* out_node  = out;                          // 25000*128
  float* out_edge  = out + (size_t)NN * 128;       // 100000*128
  float* out_cycle = out_edge + (size_t)NE2 * 128; // 88000*128

  // ---- workspace layout ----
  char* ws = (char*)d_ws;
  size_t off = 0;
  auto alloc = [&](size_t bytes) -> void* {
    void* p = ws + off;
    off = (off + bytes + 255) & ~(size_t)255;
    return p;
  };
  float* nacc1   = (float*)alloc((size_t)NN * 128 * 4);
  float* nacc2   = (float*)alloc((size_t)NN * 256 * 4);
  float* e2n     = (float*)alloc((size_t)NN * 768 * 4);
  int*   deg     = (int*)  alloc((size_t)NN * 4);
  float* ss5     = (float*)alloc((size_t)8000 * 128 * 4);
  float* ss6     = (float*)alloc((size_t)8000 * 128 * 4);
  float* naccc5  = (float*)alloc((size_t)NN * 256 * 4);
  float* naccc6  = (float*)alloc((size_t)NN * 256 * 4);
  float* ssp5    = (float*)alloc((size_t)8000 * 256 * 4);
  float* ssp6    = (float*)alloc((size_t)8000 * 256 * 4);
  float* nacc_c  = (float*)alloc((size_t)NN * 640 * 4);
  __bf16* Xee  = (__bf16*)alloc((size_t)NE2 * 896 * 2);
  __bf16* Xen  = (__bf16*)alloc((size_t)NN * 896 * 2);
  __bf16* Xec  = (__bf16*)alloc((size_t)NE2 * 1408 * 2);
  __bf16* Xcy  = (__bf16*)alloc((size_t)NCR * 640 * 2);
  __bf16* Xout = (__bf16*)alloc((size_t)NE2 * 256 * 2);
  float*  Y    = (float*) alloc((size_t)NE2 * 256 * 4);
  __bf16* Hb   = (__bf16*)alloc((size_t)NE2 * 256 * 2);
  float*  Y2   = (float*) alloc((size_t)NE2 * 128 * 4);
  float*  stats = (float*)alloc(512 * 4);  // [0:256)=colsum, [256:512)=colsumsq
  __bf16* wt_enn1 = (__bf16*)alloc((size_t)256 * 896 * 2);
  __bf16* wt_enn2 = (__bf16*)alloc((size_t)128 * 256 * 2);
  __bf16* wt_ene1 = (__bf16*)alloc((size_t)256 * 896 * 2);
  __bf16* wt_ene2 = (__bf16*)alloc((size_t)128 * 256 * 2);
  __bf16* wt_ece1 = (__bf16*)alloc((size_t)256 * 1408 * 2);
  __bf16* wt_ece2 = (__bf16*)alloc((size_t)128 * 256 * 2);
  __bf16* wt_ecc1 = (__bf16*)alloc((size_t)256 * 640 * 2);
  __bf16* wt_ecc2 = (__bf16*)alloc((size_t)128 * 256 * 2);
  __bf16* wt_out  = (__bf16*)alloc((size_t)128 * 256 * 2);

  // ---- zero the atomic accumulators ----
  hipMemsetAsync(nacc1,  0, (size_t)NN * 128 * 4, stream);
  hipMemsetAsync(nacc2,  0, (size_t)NN * 256 * 4, stream);
  hipMemsetAsync(e2n,    0, (size_t)NN * 768 * 4, stream);
  hipMemsetAsync(deg,    0, (size_t)NN * 4, stream);
  hipMemsetAsync(ss5,    0, (size_t)8000 * 128 * 4, stream);
  hipMemsetAsync(ss6,    0, (size_t)8000 * 128 * 4, stream);
  hipMemsetAsync(naccc5, 0, (size_t)NN * 256 * 4, stream);
  hipMemsetAsync(naccc6, 0, (size_t)NN * 256 * 4, stream);
  hipMemsetAsync(ssp5,   0, (size_t)8000 * 256 * 4, stream);
  hipMemsetAsync(ssp6,   0, (size_t)8000 * 256 * 4, stream);
  hipMemsetAsync(nacc_c, 0, (size_t)NN * 640 * 4, stream);

  // ---- weight transpose + bf16 convert ----
  auto wt = [&](const float* W, __bf16* Wt, int K, int N) {
    int tot = K * N;
    k_wt<<<(tot + 255) / 256, 256, 0, stream>>>(W, Wt, K, N);
  };
  wt(enn_W1, wt_enn1, 896, 256);  wt(enn_W2, wt_enn2, 256, 128);
  wt(ene_W1, wt_ene1, 896, 256);  wt(ene_W2, wt_ene2, 256, 128);
  wt(ece_W1, wt_ece1, 1408, 256); wt(ece_W2, wt_ece2, 256, 128);
  wt(ecc_W1, wt_ecc1, 640, 256);  wt(ecc_W2, wt_ecc2, 256, 128);
  wt(out_W,  wt_out,  256, 128);

  // ---- edge/node gather-scatter pipeline ----
  k_hist<<<(NE2 + 255) / 256, 256, 0, stream>>>(edge_atom, deg, NE2);
  k_scatter_rows<<<NE2, 128, 0, stream>>>(edge_rep, edge_atom, nacc1, 128);
  k_scatter_nacc2<<<NE2, 256, 0, stream>>>(nacc1, edge_atom, nacc2);
  k_scatter_e2n<<<NE2, 256, 0, stream>>>(nacc2, edge_atom, deg, node_rep, e2n);
  k_build_Xee<<<NE2, 256, 0, stream>>>(edge_rep, nacc2, edge_atom, deg, node_rep, Xee);
  k_build_Xen<<<NN, 256, 0, stream>>>(node_rep, e2n, Xen);

  // ---- cycle pipeline ----
  k_scatter_ss1<<<NC5R, 128, 0, stream>>>(nacc1, cyc_atom, ss5, 5);
  k_scatter_ss1<<<NC6R, 128, 0, stream>>>(nacc1, cyc_atom + NC5R, ss6, 6);
  k_scatter_naccc<<<NC5R, 256, 0, stream>>>(nacc1, ss5, cyc_atom, naccc5, 5);
  k_scatter_naccc<<<NC6R, 256, 0, stream>>>(nacc1, ss6, cyc_atom + NC5R, naccc6, 6);
  k_scatter_ssp<<<NC5R, 256, 0, stream>>>(naccc5, cyc_atom, ssp5, 5);
  k_scatter_ssp<<<NC6R, 256, 0, stream>>>(naccc6, cyc_atom + NC5R, ssp6, 6);
  k_build_cycle<<<NCR, 256, 0, stream>>>(naccc5, ssp5, naccc6, ssp6, cycle_rep, cyc_atom, Xcy, nacc_c);
  k_build_Xec<<<NE2, 256, 0, stream>>>(edge_rep, nacc_c, edge_atom, Xec);

  // ---- GEMM + BN helpers ----
  auto gemm = [&](const __bf16* X, const __bf16* Wt, float* Yo, int M, int N, int K) {
    hipMemsetAsync(stats, 0, 512 * 4, stream);
    dim3 grid((M + 127) / 128, N / 64);
    k_gemm_bn<<<grid, 256, 0, stream>>>(X, Wt, Yo, stats, stats + 256, M, N, K);
  };

  // MLP en_edge -> Xout[:, 0:128]
  gemm(Xee, wt_ene1, Y, NE2, 256, 896);
  k_bn_relu_bf16<<<NE2, 256, 0, stream>>>(Y, stats, stats + 256, ene_g1, ene_b1, Hb, NE2, 256, 256, 0);
  gemm(Hb, wt_ene2, Y2, NE2, 128, 256);
  k_bn_relu_bf16<<<NE2, 128, 0, stream>>>(Y2, stats, stats + 256, ene_g2, ene_b2, Xout, NE2, 128, 256, 0);

  // MLP ec_edge -> Xout[:, 128:256]
  gemm(Xec, wt_ece1, Y, NE2, 256, 1408);
  k_bn_relu_bf16<<<NE2, 256, 0, stream>>>(Y, stats, stats + 256, ece_g1, ece_b1, Hb, NE2, 256, 256, 0);
  gemm(Hb, wt_ece2, Y2, NE2, 128, 256);
  k_bn_relu_bf16<<<NE2, 128, 0, stream>>>(Y2, stats, stats + 256, ece_g2, ece_b2, Xout, NE2, 128, 256, 128);

  // out_mlp -> edge output (+ residual)
  gemm(Xout, wt_out, Y2, NE2, 128, 256);
  k_bn_relu_res<<<NE2, 128, 0, stream>>>(Y2, stats, stats + 256, out_g, out_b, edge_rep, out_edge, NE2, 128);

  // MLP en_node -> node output (+ residual)
  gemm(Xen, wt_enn1, Y, NN, 256, 896);
  k_bn_relu_bf16<<<NN, 256, 0, stream>>>(Y, stats, stats + 256, enn_g1, enn_b1, Hb, NN, 256, 256, 0);
  gemm(Hb, wt_enn2, Y2, NN, 128, 256);
  k_bn_relu_res<<<NN, 128, 0, stream>>>(Y2, stats, stats + 256, enn_g2, enn_b2, node_rep, out_node, NN, 128);

  // MLP ec_cycle -> cycle output (+ residual)
  gemm(Xcy, wt_ecc1, Y, NCR, 256, 640);
  k_bn_relu_bf16<<<NCR, 256, 0, stream>>>(Y, stats, stats + 256, ecc_g1, ecc_b1, Hb, NCR, 256, 256, 0);
  gemm(Hb, wt_ecc2, Y2, NCR, 128, 256);
  k_bn_relu_res<<<NCR, 128, 0, stream>>>(Y2, stats, stats + 256, ecc_g2, ecc_b2, cycle_rep, out_cycle, NCR, 128);
}
